// FusedMultiSpineMLP_46634754899972
// MI455X (gfx1250) — matrix operand order repaired and checked by reading, not compiled, not run on hardware
//
#include <hip/hip_runtime.h>
#include <math.h>

// ---------------------------------------------------------------------------
// FusedMultiSpineMLP for MI455X (gfx1250, wave32)
//   K0: split W_out (f32 [D,D]) -> bf16 hi/lo
//   K1: per-row RMS scale
//   K2: per-(row,spine) pipeline + 6 low-rank applies (LDS-cached params),
//       emit activations as bf16 hi/lo split
//   K3: bf16x3 WMMA GEMM (out @ W^T) with double-buffered ASYNC global->LDS
//       staging of W tiles + fused epilogue x + sigmoid(gate)*proj
// ---------------------------------------------------------------------------

typedef __bf16 bf16;
typedef __attribute__((ext_vector_type(16))) __bf16 v16bf;
typedef __attribute__((ext_vector_type(8)))  float  v8f;
typedef __attribute__((ext_vector_type(4)))  float  f32x4;
typedef __attribute__((ext_vector_type(2)))  unsigned int u32x2;
typedef __attribute__((ext_vector_type(4)))  unsigned int u32x4;

union frag16 { u32x4 q[2]; v16bf v; };
union bpack  { bf16 b[4]; u32x2 u; };

#define Ntot 16384
#define Dm   1024
#define Sn   8
#define Hh   128
#define Rr   16

__device__ __forceinline__ float wred32(float v) {
  v += __shfl_xor(v, 16, 32);
  v += __shfl_xor(v,  8, 32);
  v += __shfl_xor(v,  4, 32);
  v += __shfl_xor(v,  2, 32);
  v += __shfl_xor(v,  1, 32);
  return v;
}

// ------------------------- K0: weight split --------------------------------
__global__ __launch_bounds__(256) void split_w_kernel(const float* __restrict__ W,
                                                      bf16* __restrict__ Whi,
                                                      bf16* __restrict__ Wlo) {
  const size_t i = ((size_t)blockIdx.x * 256 + threadIdx.x) * 4;
  const f32x4 v = *(const f32x4*)(W + i);
  float f[4] = { v.x, v.y, v.z, v.w };
  bpack h, l;
#pragma unroll
  for (int j = 0; j < 4; ++j) {
    bf16 hb = (bf16)f[j];
    h.b[j] = hb;
    l.b[j] = (bf16)(f[j] - (float)hb);
  }
  *(u32x2*)(Whi + i) = h.u;
  *(u32x2*)(Wlo + i) = l.u;
}

// ------------------------- K1: RMS pass ------------------------------------
__global__ __launch_bounds__(256) void rms_kernel(const float* __restrict__ x,
                                                  float* __restrict__ rinv) {
  __shared__ float red[8];
  const int n = blockIdx.x;
  const f32x4 v = *(const f32x4*)(x + (size_t)n * Dm + threadIdx.x * 4);
  float ss = v.x * v.x + v.y * v.y + v.z * v.z + v.w * v.w;
  ss = wred32(ss);
  if ((threadIdx.x & 31) == 0) red[threadIdx.x >> 5] = ss;
  __syncthreads();
  if (threadIdx.x == 0) {
    float tot = 0.0f;
#pragma unroll
    for (int i = 0; i < 8; ++i) tot += red[i];
    rinv[n] = rsqrtf(tot * (1.0f / (float)Dm) + 1.1920929e-07f);
  }
}

// ------------------------- K2: spine pass ----------------------------------
__device__ __forceinline__ void lr_apply_lds(const float* __restrict__ Vmat,
                                             const float* __restrict__ Umat,
                                             const float in[4], int lane,
                                             float out[4]) {
  float tmp[16];
#pragma unroll
  for (int r = 0; r < 16; ++r) {
    const f32x4 vv = *(const f32x4*)&Vmat[r * Hh + lane * 4];
    tmp[r] = wred32(in[0] * vv.x + in[1] * vv.y + in[2] * vv.z + in[3] * vv.w);
  }
  out[0] = out[1] = out[2] = out[3] = 0.0f;
#pragma unroll
  for (int r = 0; r < 16; ++r) {
    const f32x4 uu = *(const f32x4*)&Umat[r * Hh + lane * 4];  // U stored transposed
    out[0] += tmp[r] * uu.x;
    out[1] += tmp[r] * uu.y;
    out[2] += tmp[r] * uu.z;
    out[3] += tmp[r] * uu.w;
  }
}

__global__ __launch_bounds__(256) void spine_kernel(
    const float* __restrict__ x, const float* __restrict__ rinv,
    const float* __restrict__ spine_proj, const float* __restrict__ P0,
    const float* __restrict__ P1,
    const float* __restrict__ U0, const float* __restrict__ U1,
    const float* __restrict__ U2, const float* __restrict__ U3,
    const float* __restrict__ U4, const float* __restrict__ U5,
    const float* __restrict__ V0, const float* __restrict__ V1,
    const float* __restrict__ V2, const float* __restrict__ V3,
    const float* __restrict__ V4, const float* __restrict__ V5,
    bf16* __restrict__ Ahi, bf16* __restrict__ Alo) {
  extern __shared__ float lds[];
  float* Um  = lds;              // 6*2048, transposed: [r*128 + d]
  float* Vm  = lds + 6 * 2048;   // 6*2048: [r*128 + d]
  float* spl = lds + 12 * 2048;  // 128
  float* p0s = spl + 128;
  float* p1s = spl + 256;

  const int tid  = threadIdx.x;
  const int lane = tid & 31;
  const int wave = tid >> 5;
  const int s    = blockIdx.y;
  const int sbase = s * (Hh * Rr);

  const float* usrc[6] = { U0, U1, U2, U3, U4, U5 };
  const float* vsrc[6] = { V0, V1, V2, V3, V4, V5 };

  for (int idx = tid; idx < 6 * 2048; idx += 256) {
    const int mtx = idx >> 11;
    const int e   = idx & 2047;
    const int d   = e >> 4;   // U global layout [H][R]: e = d*16 + r
    const int r   = e & 15;
    Um[mtx * 2048 + r * Hh + d] = usrc[mtx][sbase + e];  // transpose into LDS
    Vm[mtx * 2048 + e]          = vsrc[mtx][sbase + e];  // V already [R][H]
  }
  if (tid < 128) {
    spl[tid] = spine_proj[s * Hh + tid];
    p0s[tid] = P0[s * Hh + tid];
    p1s[tid] = P1[s * Hh + tid];
  }
  __syncthreads();

  const f32x4 sp4 = *(const f32x4*)&spl[lane * 4];
  const float nrm2 = wred32(sp4.x * sp4.x + sp4.y * sp4.y + sp4.z * sp4.z + sp4.w * sp4.w);
  const float nrm  = fmaxf(sqrtf(nrm2), 1e-12f);
  const float inv_n2 = 1.0f / (nrm * nrm);
  const f32x4 p04 = *(const f32x4*)&p0s[lane * 4];
  const f32x4 p14 = *(const f32x4*)&p1s[lane * 4];
  const float p0a[4] = { p04.x, p04.y, p04.z, p04.w };
  const float p1a[4] = { p14.x, p14.y, p14.z, p14.w };
  const float spa[4] = { sp4.x, sp4.y, sp4.z, sp4.w };

  const int n0 = blockIdx.x * 64 + wave * 8;
  for (int rr = 0; rr < 8; ++rr) {
    const int n = n0 + rr;
    const float rv = rinv[n];
    const f32x4 xv = *(const f32x4*)(x + (size_t)n * Dm + s * Hh + lane * 4);
    float rl[4];
    rl[0] = 0.5f * (tanhf(xv.x * rv) + 1.0f);
    rl[1] = 0.5f * (tanhf(xv.y * rv) + 1.0f);
    rl[2] = 0.5f * (tanhf(xv.z * rv) + 1.0f);
    rl[3] = 0.5f * (tanhf(xv.w * rv) + 1.0f);

    const float dot_rp = wred32(rl[0] * spa[0] + rl[1] * spa[1] + rl[2] * spa[2] + rl[3] * spa[3]);
    const float t_   = 1.0f / (1.0f + expf(-dot_rp));
    const float coef = dot_rp * inv_n2;

    float dl[4];
#pragma unroll
    for (int j = 0; j < 4; ++j) dl[j] = rl[j] - coef * spa[j];
    const float dn2 = wred32(dl[0] * dl[0] + dl[1] * dl[1] + dl[2] * dl[2] + dl[3] * dl[3]);
    const float dsc = rsqrtf(1.0f + dn2 * 0.25f);  // DELTA_MAX = 2
    float dt[4];
#pragma unroll
    for (int j = 0; j < 4; ++j) dt[j] = dl[j] * dsc;

    const float sigma = wred32(rl[0] + rl[1] + rl[2] + rl[3]);
    const float wden  = 1.0f / (sigma * sigma + 4.0e-4f);   // eps2 = 0.02^2
    float wv[4];
#pragma unroll
    for (int j = 0; j < 4; ++j) wv[j] = (sigma * rl[j] + 3.125e-6f) * wden;  // eps2/H

    float o0[4], o1[4], o2[4], o3[4], o4[4], o5[4];
    lr_apply_lds(Vm + 0 * 2048, Um + 0 * 2048, wv, lane, o0);  // dV0 (Pp0)
    lr_apply_lds(Vm + 1 * 2048, Um + 1 * 2048, wv, lane, o1);  // dV1 (Pp1)
    lr_apply_lds(Vm + 2 * 2048, Um + 2 * 2048, dt, lane, o2);  // D0  (E0)
    lr_apply_lds(Vm + 3 * 2048, Um + 3 * 2048, dt, lane, o3);  // D1  (E1)
    lr_apply_lds(Vm + 4 * 2048, Um + 4 * 2048, dt, lane, o4);  // T0  (Ep0)
    lr_apply_lds(Vm + 5 * 2048, Um + 5 * 2048, dt, lane, o5);  // T1  (Ep1)

    const float t2 = t_ * t_, t3 = t2 * t_;
    const float h00 =  2.0f * t3 - 3.0f * t2 + 1.0f;
    const float h01 = -2.0f * t3 + 3.0f * t2;
    const float h10 = t3 - 2.0f * t2 + t_;
    const float h11 = t3 - t2;

    bpack hv, lv;
#pragma unroll
    for (int j = 0; j < 4; ++j) {
      const float y = h00 * (p0a[j] + o2[j]) + h01 * (p1a[j] + o3[j]) +
                      h10 * (o0[j] + o4[j]) + h11 * (o1[j] + o5[j]);
      const bf16 hb = (bf16)y;
      hv.b[j] = hb;
      lv.b[j] = (bf16)(y - (float)hb);
    }
    const size_t oi = (size_t)n * Dm + s * Hh + lane * 4;
    *(u32x2*)(Ahi + oi) = hv.u;
    *(u32x2*)(Alo + oi) = lv.u;
  }
}

// ------------------------- K3: bf16x3 WMMA GEMM + epilogue -----------------
__global__ __launch_bounds__(256) void gemm_kernel(
    const bf16* __restrict__ Ahi, const bf16* __restrict__ Alo,
    const bf16* __restrict__ Whi, const bf16* __restrict__ Wlo,
    const float* __restrict__ x, const float* __restrict__ gate,
    float* __restrict__ out) {
  // double-buffered W tiles; stride 40 avoids ds_read_b128 bank conflicts
  __shared__ bf16 sBhi[2][128][40];
  __shared__ bf16 sBlo[2][128][40];

  const int tid    = threadIdx.x;
  const int lane   = tid & 31;
  const int wave   = tid >> 5;
  const int rgroup = wave & 3;
  const int cgroup = wave >> 2;
  const int rbase  = blockIdx.x * 64 + rgroup * 16;
  const int cblock = blockIdx.y * 128;
  const int cbase  = cblock + cgroup * 64;
  const int m      = lane & 15;
  const int hw     = lane >> 4;
  const int arow   = rbase + m;
  const int koffA  = hw * 8;   // A lanes: K runs {0..7,16..23} / {8..15,24..31}
  const int koffB  = hw * 8;   // B lanes: K runs {0..7,16..23} / {8..15,24..31}

  v8f acc[4] = {};

  const int wcol = tid >> 1;
  const int wkh  = (tid & 1) * 16;

  // async global->LDS staging of one W k-slab (32 k) into buffer b.
  // generic pointers to LDS carry the LDS byte offset in their low 32 bits.
  auto stage = [&](int k0, int b) {
    const bf16* gh = Whi + (size_t)(cblock + wcol) * Dm + k0 + wkh;
    const bf16* gl = Wlo + (size_t)(cblock + wcol) * Dm + k0 + wkh;
    const unsigned dh0 = (unsigned)(uintptr_t)(const void*)&sBhi[b][wcol][wkh];
    const unsigned dh1 = (unsigned)(uintptr_t)(const void*)&sBhi[b][wcol][wkh + 8];
    const unsigned dl0 = (unsigned)(uintptr_t)(const void*)&sBlo[b][wcol][wkh];
    const unsigned dl1 = (unsigned)(uintptr_t)(const void*)&sBlo[b][wcol][wkh + 8];
    const unsigned long long ah = (unsigned long long)(uintptr_t)gh;
    const unsigned long long al = (unsigned long long)(uintptr_t)gl;
    asm volatile(
        "global_load_async_to_lds_b128 %0, %4, off\n\t"
        "global_load_async_to_lds_b128 %1, %4, off offset:16\n\t"
        "global_load_async_to_lds_b128 %2, %5, off\n\t"
        "global_load_async_to_lds_b128 %3, %5, off offset:16"
        :
        : "v"(dh0), "v"(dh1), "v"(dl0), "v"(dl1), "v"(ah), "v"(al)
        : "memory");
  };

  stage(0, 0);
  asm volatile("s_wait_asynccnt 0x0" ::: "memory");
  __syncthreads();

  for (int k0 = 0; k0 < Dm; k0 += 32) {
    const int buf = (k0 >> 5) & 1;
    if (k0 + 32 < Dm) {
      stage(k0 + 32, buf ^ 1);  // prefetch next slab while computing this one
      __builtin_prefetch(Ahi + (size_t)arow * Dm + k0 + 32 + koffA, 0, 1);
      __builtin_prefetch(Alo + (size_t)arow * Dm + k0 + 32 + koffA, 0, 1);
    }

    frag16 fah, fal;
    const bf16* pah = Ahi + (size_t)arow * Dm + k0 + koffA;
    const bf16* pal = Alo + (size_t)arow * Dm + k0 + koffA;
    fah.q[0] = *(const u32x4*)pah;
    fah.q[1] = *(const u32x4*)(pah + 16);
    fal.q[0] = *(const u32x4*)pal;
    fal.q[1] = *(const u32x4*)(pal + 16);

#pragma unroll
    for (int ct = 0; ct < 4; ++ct) {
      const int lcol = cgroup * 64 + ct * 16 + m;
      frag16 bh, bl;
      bh.q[0] = *(const u32x4*)&sBhi[buf][lcol][koffB];
      bh.q[1] = *(const u32x4*)&sBhi[buf][lcol][koffB + 16];
      bl.q[0] = *(const u32x4*)&sBlo[buf][lcol][koffB];
      bl.q[1] = *(const u32x4*)&sBlo[buf][lcol][koffB + 16];
      acc[ct] = __builtin_amdgcn_wmma_f32_16x16x32_bf16(false, fah.v, false, bh.v,
                                                        (short)0, acc[ct], false, false);
      acc[ct] = __builtin_amdgcn_wmma_f32_16x16x32_bf16(false, fah.v, false, bl.v,
                                                        (short)0, acc[ct], false, false);
      acc[ct] = __builtin_amdgcn_wmma_f32_16x16x32_bf16(false, fal.v, false, bh.v,
                                                        (short)0, acc[ct], false, false);
    }

    // my async loads for the next slab have landed; barrier makes the whole
    // tile (all waves' contributions) visible before the next iteration.
    asm volatile("s_wait_asynccnt 0x0" ::: "memory");
    __syncthreads();
  }

  const float g = 1.0f / (1.0f + expf(-gate[0]));
#pragma unroll
  for (int ct = 0; ct < 4; ++ct) {
    const int col = cbase + ct * 16 + m;
#pragma unroll
    for (int i = 0; i < 8; ++i) {  // C/D layout: VGPR i -> M = i + 8*hw, N = m
      const int row = rbase + i + hw * 8;
      const size_t idx = (size_t)row * Dm + col;
      out[idx] = x[idx] + g * acc[ct][i];
    }
  }
}

// ------------------------- launch ------------------------------------------
extern "C" void kernel_launch(void* const* d_in, const int* in_sizes, int n_in,
                              void* d_out, int out_size, void* d_ws, size_t ws_size,
                              hipStream_t stream) {
  (void)in_sizes; (void)n_in; (void)out_size; (void)ws_size;

  const float* x          = (const float*)d_in[0];
  const float* spine_proj = (const float*)d_in[1];
  const float* P0         = (const float*)d_in[2];
  const float* P1         = (const float*)d_in[3];
  const float* Pp0_U = (const float*)d_in[4],  *Pp0_V = (const float*)d_in[5];
  const float* Pp1_U = (const float*)d_in[6],  *Pp1_V = (const float*)d_in[7];
  const float* E0_U  = (const float*)d_in[8],  *E0_V  = (const float*)d_in[9];
  const float* E1_U  = (const float*)d_in[10], *E1_V  = (const float*)d_in[11];
  const float* Ep0_U = (const float*)d_in[12], *Ep0_V = (const float*)d_in[13];
  const float* Ep1_U = (const float*)d_in[14], *Ep1_V = (const float*)d_in[15];
  const float* W_out = (const float*)d_in[16];
  const float* gate  = (const float*)d_in[17];
  float* out = (float*)d_out;

  // workspace carve-up
  bf16* Ahi = (bf16*)d_ws;
  bf16* Alo = Ahi + (size_t)Ntot * Dm;
  bf16* Whi = Alo + (size_t)Ntot * Dm;
  bf16* Wlo = Whi + (size_t)Dm * Dm;
  float* rinv = (float*)(Wlo + (size_t)Dm * Dm);

  split_w_kernel<<<(Dm * Dm) / (256 * 4), 256, 0, stream>>>(W_out, Whi, Wlo);
  rms_kernel<<<Ntot, 256, 0, stream>>>(x, rinv);

  dim3 g2(Ntot / 64, Sn);
  const size_t lds_bytes = (size_t)(12 * 2048 + 3 * 128) * sizeof(float);  // 99,840 B
  spine_kernel<<<g2, 256, lds_bytes, stream>>>(
      x, rinv, spine_proj, P0, P1,
      Pp0_U, Pp1_U, E0_U, E1_U, Ep0_U, Ep1_U,
      Pp0_V, Pp1_V, E0_V, E1_V, Ep0_V, Ep1_V,
      Ahi, Alo);

  dim3 g3(Ntot / 64, Dm / 128);
  gemm_kernel<<<g3, 256, 0, stream>>>(Ahi, Alo, Whi, Wlo, x, gate, out);
}
